// sMLP4_49160195670410
// MI455X (gfx1250) — compile-verified, matrix operand order checked by reading, not run on hardware
//
#include <hip/hip_runtime.h>
#include <hip/hip_bf16.h>
#include <stdint.h>

// ---------------------------------------------------------------------------
// Problem constants (from the reference)
// ---------------------------------------------------------------------------
#define B_SZ      1024
#define IN_DIM    784
#define K1PAD     800      // 784 padded to multiple of 32
#define HID       1000
#define N1PAD     1024     // 1000 padded (also = K of GEMM2)
#define OUT2      100
#define N2PAD     112      // 100 padded to multiple of 16
#define NSTEPS    50

// Toolchain-dependent arity of the TDM builtin (see CDNA5_HIP.md):
// therock headers ship amd_gfx1250_TDM.h and use the 6-arg form.
#if defined(__has_include)
#  if __has_include(<hip/amd_detail/amd_gfx1250_TDM.h>)
#    define TDM_SIX_ARGS 1
#  else
#    define TDM_SIX_ARGS 0
#  endif
#else
#  define TDM_SIX_ARGS 0
#endif

typedef __attribute__((ext_vector_type(16))) __bf16 v16bf;
typedef __attribute__((ext_vector_type(8)))  float  v8f;
typedef __attribute__((ext_vector_type(4)))  unsigned int u32x4;
typedef __attribute__((ext_vector_type(8)))  int    i32x8;
typedef __attribute__((ext_vector_type(4)))  int    i32x4;

union Frag16 {           // one 16x32 bf16 WMMA operand fragment (per lane: 32B)
  v16bf v;
  uint4 u[2];
};

// lane l (0..31): row/col = l&15 ; K elements [koff, koff+8) and [koff+16, koff+24)
// where koff = (l>>4)*8  -> matches CDNA5 16-bit A/B VGPR layout (ISA 7.12.2)
__device__ __forceinline__ v16bf load_frag_bf16(const unsigned short* p) {
  Frag16 f;
  f.u[0] = *reinterpret_cast<const uint4*>(p);        // K = koff .. koff+7
  f.u[1] = *reinterpret_cast<const uint4*>(p + 16);   // K = koff+16 .. koff+23
  return f.v;
}

__device__ __forceinline__ unsigned short f32_to_bf16(float f) {
  union { float f; uint32_t u; } v; v.f = f;
  uint32_t r = v.u + 0x7FFFu + ((v.u >> 16) & 1u);    // round-to-nearest-even
  return (unsigned short)(r >> 16);
}

// Counter-based uniform RNG in [0,1): stand-in for jax threefry stream
__device__ __forceinline__ float urand(uint32_t idx) {
  uint32_t h = idx * 0x9E3779B9u + 0x85EBCA6Bu;
  h ^= h >> 16; h *= 0x85EBCA6Bu;
  h ^= h >> 13; h *= 0xC2B2AE35u;
  h ^= h >> 16;
  return (float)(h >> 8) * (1.0f / 16777216.0f);
}

// Approximate sigmoid (decay constant): v_exp + v_rcp, no precise-div chain
__device__ __forceinline__ float dsigmoid(float t) {
  return __builtin_amdgcn_rcpf(1.0f + __expf(-t));
}

// ---------------------------------------------------------------------------
// TDM: async 2D tile load (64 rows x 32 bf16 elems) global -> LDS.
// D# per CDNA5 ISA: group0 = {count|flags, lds_addr,
// global_addr, type=2}; group1 = {data_size=1(2B), tensor_dim0/1 = 1024,
// tile_dim0=32, tile_dim1=64, tensor_dim0_stride=1024}. Groups 2/3 zero
// (tile_dim2..4 = 0 -> unused).
// ---------------------------------------------------------------------------
__device__ __forceinline__ void tdm_load_tile_64x32(uint32_t lds_off,
                                                    const unsigned short* gsrc) {
  uint64_t ga = (uint64_t)(uintptr_t)gsrc;
  u32x4 g0;
  g0.x = 1u;                                                  // count=1, user D#
  g0.y = lds_off;                                             // LDS byte address
  g0.z = (uint32_t)ga;                                        // global_addr[31:0]
  g0.w = (uint32_t)((ga >> 32) & 0x01FFFFFFu) | (2u << 30);   // [56:32] | type=2
  i32x8 g1;
  g1[0] = 0x00010000;                         // data_size=1 (2 bytes/elem)
  g1[1] = (int)((N1PAD & 0xFFFFu) << 16);     // tensor_dim0[15:0] in [31:16]
  g1[2] = (int)((N1PAD >> 16) | ((N1PAD & 0xFFFFu) << 16)); // td0 hi | td1 lo
  g1[3] = (int)((N1PAD >> 16) | (32u << 16)); // td1 hi | tile_dim0=32
  g1[4] = 64;                                 // tile_dim1=64, tile_dim2=0
  g1[5] = N1PAD;                              // tensor_dim0_stride[31:0]
  g1[6] = 0;                                  // stride hi | dim1_stride lo
  g1[7] = 0;
#if TDM_SIX_ARGS
  __builtin_amdgcn_tensor_load_to_lds(g0, g1, (i32x4)0, (i32x4)0, (i32x8)0, 0);
#else
  __builtin_amdgcn_tensor_load_to_lds(g0, g1, (i32x4)0, (i32x4)0, 0);
#endif
}

// ---------------------------------------------------------------------------
// One-time weight conversion to padded bf16 (zero padding keeps math exact)
// ---------------------------------------------------------------------------
__global__ void conv_w1(const float* __restrict__ w1, unsigned short* __restrict__ w1b) {
  int idx = blockIdx.x * blockDim.x + threadIdx.x;            // over N1PAD*K1PAD
  if (idx >= N1PAD * K1PAD) return;
  int n = idx / K1PAD, k = idx % K1PAD;
  float v = (n < HID && k < IN_DIM) ? w1[n * IN_DIM + k] : 0.0f;
  w1b[idx] = f32_to_bf16(v);
}

__global__ void conv_w2(const float* __restrict__ w2, unsigned short* __restrict__ w2b) {
  int idx = blockIdx.x * blockDim.x + threadIdx.x;            // over N2PAD*N1PAD
  if (idx >= N2PAD * N1PAD) return;
  int n = idx / N1PAD, k = idx % N1PAD;
  float v = (n < OUT2 && k < HID) ? w2[n * HID + k] : 0.0f;
  w2b[idx] = f32_to_bf16(v);
}

// ---------------------------------------------------------------------------
// Per-step Bernoulli spike generation, written as bf16 {0,1} (exact)
// ---------------------------------------------------------------------------
__global__ void spikegen(const float* __restrict__ x, unsigned short* __restrict__ spk, int t) {
  int idx = blockIdx.x * blockDim.x + threadIdx.x;            // over B_SZ*K1PAD
  if (idx >= B_SZ * K1PAD) return;
  int b = idx / K1PAD, k = idx % K1PAD;
  unsigned short s = 0;
  if (k < IN_DIM) {
    float u = urand((uint32_t)t * (uint32_t)(B_SZ * IN_DIM) + (uint32_t)(b * IN_DIM + k));
    s = (u < x[b * IN_DIM + k]) ? (unsigned short)0x3F80 : (unsigned short)0;
  }
  spk[idx] = s;
}

// ---------------------------------------------------------------------------
// GEMM1 + LIF:  I = spk @ W1^T ;  h1m = h1m_old*d0*[h1m_old<1] + I ; h1s = [h1m>=1]
// Block: 128 thr = 4 waves (2M x 2N); each wave -> 32x64 tile (2x4 WMMA accums:
// 12 b128 loads per 8 WMMAs). Grid: (1024/64, 1024/128) = (16, 8)
// ---------------------------------------------------------------------------
__global__ __launch_bounds__(128) void gemm1_lif(
    const unsigned short* __restrict__ spk,   // [B_SZ][K1PAD] bf16
    const unsigned short* __restrict__ w1b,   // [N1PAD][K1PAD] bf16
    float* __restrict__ h1m,                  // [B_SZ][N1PAD] f32 state
    unsigned short* __restrict__ h1s,         // [B_SZ][N1PAD] bf16 spikes
    const float* __restrict__ tau0, int t)
{
  const int wave = threadIdx.x >> 5;
  const int lane = threadIdx.x & 31;
  const int wm = wave & 1, wn = wave >> 1;
  const int m0 = blockIdx.x * 64 + wm * 32;
  const int n0 = blockIdx.y * 128 + wn * 64;
  const int rc   = lane & 15;         // row (A) / col (B) within a 16-tile
  const int koff = (lane >> 4) * 8;   // K-half select

  v8f acc[2][4] = {};
  for (int kk = 0; kk < K1PAD; kk += 32) {
    v16bf a[2], b[4];
#pragma unroll
    for (int i = 0; i < 2; ++i)
      a[i] = load_frag_bf16(spk + (size_t)(m0 + 16 * i + rc) * K1PAD + kk + koff);
#pragma unroll
    for (int j = 0; j < 4; ++j)
      b[j] = load_frag_bf16(w1b + (size_t)(n0 + 16 * j + rc) * K1PAD + kk + koff);
#pragma unroll
    for (int i = 0; i < 2; ++i)
#pragma unroll
      for (int j = 0; j < 4; ++j)
        acc[i][j] = __builtin_amdgcn_wmma_f32_16x16x32_bf16(
            false, a[i], false, b[j], (short)0, acc[i][j], false, false);
  }

  const float d0 = dsigmoid(tau0[0]);
  const int rbase = 8 * (lane >> 4);
#pragma unroll
  for (int i = 0; i < 2; ++i)
#pragma unroll
    for (int j = 0; j < 4; ++j)
#pragma unroll
      for (int v = 0; v < 8; ++v) {
        int m = m0 + 16 * i + rbase + v;      // C/D layout: VGPR v -> row
        int n = n0 + 16 * j + rc;             //             lane   -> col
        size_t o = (size_t)m * N1PAD + n;
        float old = (t == 0) ? 0.5f : h1m[o];
        float mem = ((old < 1.0f) ? old * d0 : 0.0f) + acc[i][j][v];
        h1m[o] = mem;
        h1s[o] = (mem >= 1.0f) ? (unsigned short)0x3F80 : (unsigned short)0;
      }
}

// ---------------------------------------------------------------------------
// GEMM2 + LIF + AvgPool(10,10) + leaky accumulator
// Block: 224 thr = 7 waves; wave w -> N-tile w (16 cols), 64 rows (4 accums,
// B fragment shared across the 4 M-tiles). The A-tile (h1s, 64x32 bf16 = 4KB)
// is identical for all 7 waves -> staged via TDM tensor_load_to_lds, double
// buffered (issue chunk i+1 while computing chunk i). Grid: 1024/64 = 16.
// ---------------------------------------------------------------------------
__global__ __launch_bounds__(224) void gemm2_lif_pool(
    const unsigned short* __restrict__ h1s,   // [B_SZ][N1PAD] bf16
    const unsigned short* __restrict__ w2b,   // [N2PAD][N1PAD] bf16
    float* __restrict__ h2m,                  // [B_SZ][N2PAD] f32 state
    float* __restrict__ out,                  // [B_SZ][10] f32 (acc state = output)
    const float* __restrict__ tauv,
    const float* __restrict__ acct, int t)
{
  __shared__ __align__(16) unsigned short s_stage[2][64][32];  // 2 x 4 KB
  __shared__ __align__(16) float s_spk[64][N2PAD];             // 28 KB

  const int wave = threadIdx.x >> 5;          // 0..6
  const int lane = threadIdx.x & 31;
  const int n0 = wave * 16;
  const int m0 = blockIdx.x * 64;
  const int rc   = lane & 15;
  const int koff = (lane >> 4) * 8;

  const unsigned short* gA = h1s + (size_t)m0 * N1PAD;
  const uint32_t lds0 = (uint32_t)(uintptr_t)&s_stage[0][0][0];
  const uint32_t lds1 = (uint32_t)(uintptr_t)&s_stage[1][0][0];

  if (wave == 0) tdm_load_tile_64x32(lds0, gA);               // prime chunk 0

  v8f acc[4] = {};
  const int NIT = N1PAD / 32;                                  // 32 K-chunks
  for (int it = 0; it < NIT; ++it) {
    if (wave == 0) __builtin_amdgcn_s_wait_tensorcnt(0);       // chunk it landed
    __syncthreads();                                           // visible; buf free
    if (wave == 0 && it + 1 < NIT)
      tdm_load_tile_64x32((it & 1) ? lds0 : lds1, gA + (size_t)(it + 1) * 32);

    const unsigned short* tile = &s_stage[it & 1][0][0];
    const int kk = it * 32;
    v16bf b = load_frag_bf16(w2b + (size_t)(n0 + rc) * N1PAD + kk + koff);
#pragma unroll
    for (int i = 0; i < 4; ++i) {
      v16bf a = load_frag_bf16(tile + (16 * i + rc) * 32 + koff);  // ds_load_b128
      acc[i] = __builtin_amdgcn_wmma_f32_16x16x32_bf16(
          false, a, false, b, (short)0, acc[i], false, false);
    }
  }

  const float dv = dsigmoid(tauv[0]);
  const int rbase = 8 * (lane >> 4);
#pragma unroll
  for (int i = 0; i < 4; ++i)
#pragma unroll
    for (int v = 0; v < 8; ++v) {
      int m = m0 + 16 * i + rbase + v;
      int n = n0 + rc;
      size_t o = (size_t)m * N2PAD + n;
      float old = (t == 0) ? 0.5f : h2m[o];
      float mem = ((old < 1.0f) ? old * dv : 0.0f) + acc[i][v];
      h2m[o] = mem;
      s_spk[m - m0][n] = (mem >= 1.0f) ? 1.0f : 0.0f;
    }
  __syncthreads();

  const float da = dsigmoid(acct[0]);
  for (int idx = threadIdx.x; idx < 64 * 10; idx += 224) {
    int r = idx / 10, j = idx % 10;
    float s = 0.0f;
#pragma unroll
    for (int c = 0; c < 10; ++c) s += s_spk[r][10 * j + c];
    float boost = s * 0.1f;
    size_t gi = (size_t)(m0 + r) * 10 + j;
    out[gi] = ((t == 0) ? 0.0f : out[gi] * da) + boost;
  }
}

// ---------------------------------------------------------------------------
// Host-side orchestration (graph-capture safe: launches only)
// ---------------------------------------------------------------------------
extern "C" void kernel_launch(void* const* d_in, const int* in_sizes, int n_in,
                              void* d_out, int out_size, void* d_ws, size_t ws_size,
                              hipStream_t stream) {
  const float* x    = (const float*)d_in[0];
  const float* W1   = (const float*)d_in[1];
  const float* W2   = (const float*)d_in[2];
  const float* tau0 = (const float*)d_in[3];
  const float* tauv = (const float*)d_in[4];
  const float* acct = (const float*)d_in[5];
  float* out = (float*)d_out;

  // Workspace carve-up (all 16B aligned by construction)
  char* p = (char*)d_ws;
  unsigned short* w1b = (unsigned short*)p;  p += (size_t)N1PAD * K1PAD * 2;   // 1.6 MB
  unsigned short* w2b = (unsigned short*)p;  p += (size_t)N2PAD * N1PAD * 2;   // 0.22 MB
  unsigned short* spk = (unsigned short*)p;  p += (size_t)B_SZ  * K1PAD * 2;   // 1.6 MB
  unsigned short* h1s = (unsigned short*)p;  p += (size_t)B_SZ  * N1PAD * 2;   // 2.1 MB
  float*          h1m = (float*)p;           p += (size_t)B_SZ  * N1PAD * 4;   // 4.2 MB
  float*          h2m = (float*)p;           /* p += B_SZ*N2PAD*4 */           // 0.46 MB

  // One-time bf16 weight conversion (re-run every call: deterministic)
  {
    int n1 = N1PAD * K1PAD;
    conv_w1<<<(n1 + 255) / 256, 256, 0, stream>>>(W1, w1b);
    int n2 = N2PAD * N1PAD;
    conv_w2<<<(n2 + 255) / 256, 256, 0, stream>>>(W2, w2b);
  }

  dim3 g1(B_SZ / 64, N1PAD / 128);   // (16, 8)
  for (int t = 0; t < NSTEPS; ++t) {
    int ns = B_SZ * K1PAD;
    spikegen<<<(ns + 255) / 256, 256, 0, stream>>>(x, spk, t);
    gemm1_lif<<<g1, 128, 0, stream>>>(spk, w1b, h1m, h1s, tau0, t);
    gemm2_lif_pool<<<B_SZ / 64, 224, 0, stream>>>(h1s, w2b, h2m, out, tauv, acct, t);
  }
}
